// SE3EquivariantModel_420906795768
// MI455X (gfx1250) — matrix operand order repaired and checked by reading, not compiled, not run on hardware
//
#include <hip/hip_runtime.h>
#include <hip/hip_bf16.h>
#include <math.h>

// ---------------------------------------------------------------------------
// MI455X (gfx1250) implementation of the SE(3) GNN reference.
//
// The per-edge MLP (524288 edges x (4->128 silu -> 128x128)) is ~52 GFLOP and
// dominates; it is fused (edge features + MLP via v_wmma_f32_16x16x32_f16 +
// sigmoid + weighted aggregation) so the 256MB/layer edge tensor is never
// materialized. Partial sums per r-split go to workspace and are reduced
// deterministically (no float atomics).
//
// Round-3 changes (driven by round-2 disasm):
//  * one ligand row per wave (was 2): round 2 interleaved both rows' WMMA
//    chains -> doubled live accumulators -> scratch spills. Grid now 32x16.
//  * WMMA accumulator chains start from inline C=0; the b2 bias is added in
//    the sigmoid step from registers hoisted once per kernel (removes the
//    v_dual_mov bias-splat storm and ds bias loads per tile).
//  * W2 fragment pre-swizzle packs 4 halfs per ds_store_b64.
//  * __launch_bounds__(256, 6): ~110 live VGPRs -> higher occupancy to hide
//    v_exp/v_rcp latency; 16-bit WMMA issues as TRANS and co-executes with
//    the silu VALU stream.
//
// Input order assumption (insertion-order pytree flatten of setup_inputs):
//  0 h_l(256x32) 1 R_t(3x3) 2 p_t(3) 3 h_r(2048x32) 4 x_r(2048x3) 5 t(1)
//  6 embed_W 7 embed_b 8 time_W1 9 time_b1 10 time_W2 11 time_b2
//  12+10*L: edge_W1, edge_b1, edge_W2, edge_b2, upd_W1, upd_b1, ln_g, ln_b,
//           upd_W2, upd_b2      (L = 0,1,2)
//  42 v_W1 43 v_b1 44 v_W2 45 v_b2 46 w_W1 47 w_b1 48 w_W2 49 w_b2
// ---------------------------------------------------------------------------

typedef __attribute__((ext_vector_type(16))) _Float16 v16h;
typedef __attribute__((ext_vector_type(4)))  _Float16 v4h;
typedef __attribute__((ext_vector_type(8)))  float    v8f;

#define N_L     256
#define N_R     2048
#define HID     128
#define NSPLIT  16                   // r-splits per layer kernel
#define R_PER_SPLIT (N_R / NSPLIT)   // 128
#define CHUNKS  (R_PER_SPLIT / 16)   // 8 r-chunks of 16

__device__ __forceinline__ float fast_rcp(float x) {
    return __builtin_amdgcn_rcpf(x);          // v_rcp_f32
}
__device__ __forceinline__ float sigm_f(float x) {
    return fast_rcp(1.0f + __expf(-x));
}
__device__ __forceinline__ float silu_f(float x) {
    return x * sigm_f(x);
}

// ---------------------------------------------------------------------------
// K0: time MLP  t -> silu(t*W1+b1) @ W2 + b2   (1 block, 128 threads)
// ---------------------------------------------------------------------------
__global__ void time_mlp_kernel(const float* __restrict__ t,
                                const float* __restrict__ tW1,
                                const float* __restrict__ tb1,
                                const float* __restrict__ tW2,
                                const float* __restrict__ tb2,
                                float* __restrict__ t_emb) {
    __shared__ float hid[HID];
    int j = threadIdx.x;
    float tt = t[0];
    hid[j] = silu_f(tt * tW1[j] + tb1[j]);
    __syncthreads();
    float e = tb2[j];
    for (int k = 0; k < HID; ++k) e += hid[k] * tW2[k * HID + j];
    t_emb[j] = e;
}

// ---------------------------------------------------------------------------
// K1: embed (rows x 32) @ (32x128) + b (+ optional t_emb)
// ---------------------------------------------------------------------------
__global__ void embed_kernel(const float* __restrict__ hin,
                             const float* __restrict__ eW,
                             const float* __restrict__ eb,
                             const float* __restrict__ t_emb, // may be null
                             float* __restrict__ hout) {
    int idx = blockIdx.x * blockDim.x + threadIdx.x;
    int row = idx >> 7;
    int c   = idx & 127;
    const float* hr = hin + (size_t)row * 32;
    float acc = eb[c];
#pragma unroll
    for (int k = 0; k < 32; ++k) acc += hr[k] * eW[k * HID + c];
    if (t_emb) acc += t_emb[c];
    hout[(size_t)row * HID + c] = acc;
}

// ---------------------------------------------------------------------------
// K2: x_l = h_l[:, :3] @ R^T + p   (1 block, 256 threads, one l per thread)
// ---------------------------------------------------------------------------
__global__ void xl_kernel(const float* __restrict__ h_l,
                          const float* __restrict__ R,
                          const float* __restrict__ p,
                          float* __restrict__ x_l) {
    int l = threadIdx.x;
    float h0 = h_l[(size_t)l * HID + 0];
    float h1 = h_l[(size_t)l * HID + 1];
    float h2 = h_l[(size_t)l * HID + 2];
#pragma unroll
    for (int d = 0; d < 3; ++d)
        x_l[l * 3 + d] = h0 * R[d * 3 + 0] + h1 * R[d * 3 + 1] + h2 * R[d * 3 + 2] + p[d];
}

// ---------------------------------------------------------------------------
// K3: fused edge MLP + sigmoid + weighted aggregation (the hot kernel).
// grid = (32 l-tiles of 8, NSPLIT r-splits), block = 256 threads (8 waves).
// One ligand row per wave; per 16-r chunk the wave builds its A fragment
// (16 edges x 32 K of silu(edge_raw@W1+b1)) in registers in the CDNA5 A
// layout and runs 8 Ntiles x 4 Ksteps of v_wmma_f32_16x16x32_f16 against
// edge_W2 fragments pre-swizzled into LDS.
// ---------------------------------------------------------------------------
__global__ void __launch_bounds__(256, 6)
edge_agg_kernel(const float* __restrict__ h_l,
                const float* __restrict__ x_l,
                const float* __restrict__ h_r,
                const float* __restrict__ x_r,
                const float* __restrict__ eW1,   // 4 x 128
                const float* __restrict__ eb1,
                const float* __restrict__ eW2,   // 128 x 128
                const float* __restrict__ eb2,
                float* __restrict__ out_num,     // [NSPLIT][256][128]
                float* __restrict__ out_den) {
    __shared__ __align__(32) _Float16 w2frag[8][4][32][16]; // B fragments, 32 KB
    __shared__ __align__(16) float w1p[HID][4];  // {W1[0,k],W1[1,k],W1[2,k],W1[3,k]+b1[k]}
    __shared__ __align__(16) float hr_tile[16][HID];
    __shared__ float xl_s[8][3], lp_s[8][3];
    __shared__ float xr_s[16][3], rp_s[16][3];

    const int tid    = threadIdx.x;
    const int ltile8 = blockIdx.x;       // 8 ligand rows per block
    const int split  = blockIdx.y;
    const int wv     = tid >> 5;
    const int lane   = tid & 31;
    const int hi     = lane >> 4;        // which half of the wave
    const int rl     = lane & 15;        // A-row (r index) / C-column (n index)

    // ---- one-time LDS init ----
    if (tid < HID) {
        int k = tid;
        w1p[k][0] = eW1[k];
        w1p[k][1] = eW1[HID + k];
        w1p[k][2] = eW1[2 * HID + k];
        w1p[k][3] = eW1[3 * HID + k] + eb1[k];   // fold "+1*W1[3]" and b1
    }
    // Pre-swizzle W2 (f32 -> f16) into B-fragment layout, 4 halfs per store:
    // fragment (nt,kt): lane holds column n = nt*16 + (lane&15),
    // K = kt*32 + 16*(lane>>4) + i  for i = 0..15 (contiguous halfs).
    for (int it = 0; it < 16; ++it) {
        int idx = tid + it * 256;        // over 4096 v4h groups
        int g   = idx & 3;               // i-group: i = 4g..4g+3
        int ln  = (idx >> 2) & 31;
        int kt  = (idx >> 7) & 3;
        int nt  = (idx >> 9) & 7;
        int kb  = kt * 32 + ((ln >> 4) << 4) + g * 4;
        int n   = nt * 16 + (ln & 15);
        v4h pk;
#pragma unroll
        for (int j = 0; j < 4; ++j) pk[j] = (_Float16)eW2[(kb + j) * HID + n];
        *(v4h*)&w2frag[nt][kt][ln][g * 4] = pk;  // ds_store_b64
    }
    // ligand tile positions & props (h_l[:, -3:])
    if (tid < 8) {
        int lg = ltile8 * 8 + tid;
#pragma unroll
        for (int d = 0; d < 3; ++d) {
            xl_s[tid][d] = x_l[lg * 3 + d];
            lp_s[tid][d] = h_l[(size_t)lg * HID + (HID - 3) + d];
        }
    }
    __syncthreads();

    // wave-constant ligand row data + per-lane output-column biases
    float xl0 = xl_s[wv][0], xl1 = xl_s[wv][1], xl2 = xl_s[wv][2];
    float lp0 = lp_s[wv][0], lp1 = lp_s[wv][1], lp2 = lp_s[wv][2];
    float bbr[8];
#pragma unroll
    for (int nt = 0; nt < 8; ++nt) bbr[nt] = eb2[nt * 16 + rl];

    float num[8], den[8];
#pragma unroll
    for (int b = 0; b < 8; ++b) { num[b] = 0.0f; den[b] = 0.0f; }

    const int r_base = split * R_PER_SPLIT;

    for (int chunk = 0; chunk < CHUNKS; ++chunk) {
        const int r0 = r_base + chunk * 16;
        __syncthreads();            // previous tile fully consumed
        // stage h_r tile (16 x 128) with float4 traffic + r props + x_r
        {
            const float4* src = (const float4*)(h_r + (size_t)r0 * HID);
            float4* dst = (float4*)&hr_tile[0][0];
#pragma unroll
            for (int i = 0; i < 2; ++i) dst[tid + i * 256] = src[tid + i * 256];
        }
        if (tid < 16) {
#pragma unroll
            for (int d = 0; d < 3; ++d) {
                xr_s[tid][d] = x_r[(r0 + tid) * 3 + d];
                rp_s[tid][d] = h_r[(size_t)(r0 + tid) * HID + (HID - 3) + d];
            }
        }
        // prefetch next chunk's h_r rows (global_prefetch_b8 path)
        if (chunk + 1 < CHUNKS && tid < 16)
            __builtin_prefetch(&h_r[(size_t)(r0 + 16 + tid) * HID], 0, 1);
        __syncthreads();

        // per-lane edge scalars: this lane supplies A-row m = rl (r_local)
        float dx = xl0 - xr_s[rl][0];
        float dy = xl1 - xr_s[rl][1];
        float dz = xl2 - xr_s[rl][2];
        float d2 = dx * dx + dy * dy + dz * dz;
        float df = __expf(-d2 * 0.2f);                       // exp(-dist^2/5)
        float hb = lp0 * rp_s[rl][1] + lp1 * rp_s[rl][0];
        float hy = lp2 * rp_s[rl][2];

        // A fragments: hidden = silu(edge_raw @ W1 + b1), contiguous-K A
        // layout (same as the W2 pre-swizzle): element i of lane (m=rl, hi) holds
        // K = kt*32 + 16*hi + i.
        v16h a[4];
#pragma unroll
        for (int kt = 0; kt < 4; ++kt) {
            v16h av;
#pragma unroll
            for (int i = 0; i < 16; ++i) {
                int koff = 16 * hi + i;
                int k = kt * 32 + koff;
                float4 w = *(const float4*)&w1p[k][0];   // one ds_load_b128
                float pre = df * w.x + hb * w.y + hy * w.z + w.w;
                av[i] = (_Float16)silu_f(pre);
            }
            a[kt] = av;
        }

        // e = hidden @ W2 (+b2 in sigmoid); then weighted aggregation.
#pragma unroll
        for (int nt = 0; nt < 8; ++nt) {
            const int c = nt * 16 + rl;          // C/D column for this lane
            v8f acc = {};                        // inline C = 0
#pragma unroll
            for (int kt = 0; kt < 4; ++kt) {
                v16h bfrag = *(const v16h*)&w2frag[nt][kt][lane][0];
                acc = __builtin_amdgcn_wmma_f32_16x16x32_f16(
                    false, a[kt], false, bfrag, (short)0, acc, false, false);
            }
            // C layout: VGPR j holds row M = j + 8*hi (edge r_local = M)
            const float bb = bbr[nt];
#pragma unroll
            for (int j = 0; j < 8; ++j) {
                float w = sigm_f(acc[j] + bb);
                num[nt] += w * hr_tile[j + 8 * hi][c];
                den[nt] += w;
            }
        }
    }

    // combine the two half-wave row groups and emit per-split partials
    const int lg = ltile8 * 8 + wv;
#pragma unroll
    for (int nt = 0; nt < 8; ++nt) {
        float nv = num[nt] + __shfl_xor(num[nt], 16, 32);
        float dv = den[nt] + __shfl_xor(den[nt], 16, 32);
        if (hi == 0) {
            size_t o = (((size_t)split * N_L + lg) << 7) + (nt * 16 + rl);
            out_num[o] = nv;
            out_den[o] = dv;
        }
    }
}

// ---------------------------------------------------------------------------
// K4: reduce split partials -> agg, then node update MLP + LayerNorm + SiLU
// + residual. One block per ligand row, 128 threads (one channel each).
// ---------------------------------------------------------------------------
__global__ void __launch_bounds__(128)
update_kernel(float* __restrict__ h_l,
              const float* __restrict__ pn,
              const float* __restrict__ pd,
              const float* __restrict__ uW1,   // 256 x 128
              const float* __restrict__ ub1,
              const float* __restrict__ lng,
              const float* __restrict__ lnb,
              const float* __restrict__ uW2,   // 128 x 128
              const float* __restrict__ ub2) {
    __shared__ float hl_s[HID], agg_s[HID], s_s[HID], red[HID];
    const int l = blockIdx.x;
    const int c = threadIdx.x;

    float numv = 0.0f, denv = 0.0f;
    for (int s = 0; s < NSPLIT; ++s) {
        size_t o = (((size_t)s * N_L + l) << 7) + c;
        numv += pn[o];
        denv += pd[o];
    }
    float agg = numv / (denv + 1e-6f);   // precise div (once per element)

    hl_s[c]  = h_l[(size_t)l * HID + c];
    agg_s[c] = agg;
    __syncthreads();

    float z = ub1[c];
    for (int j = 0; j < HID; ++j) z += hl_s[j]  * uW1[j * HID + c];
    for (int j = 0; j < HID; ++j) z += agg_s[j] * uW1[(HID + j) * HID + c];

    // LayerNorm (population mean / var over the 128 channels)
    red[c] = z; __syncthreads();
    for (int off = 64; off > 0; off >>= 1) {
        if (c < off) red[c] += red[c + off];
        __syncthreads();
    }
    float mean = red[0] * (1.0f / HID);
    __syncthreads();
    float dv = z - mean;
    red[c] = dv * dv; __syncthreads();
    for (int off = 64; off > 0; off >>= 1) {
        if (c < off) red[c] += red[c + off];
        __syncthreads();
    }
    float var = red[0] * (1.0f / HID);
    __syncthreads();

    float zn = dv * rsqrtf(var + 1e-5f) * lng[c] + lnb[c];
    s_s[c] = silu_f(zn);
    __syncthreads();

    float z2 = ub2[c];
    for (int j = 0; j < HID; ++j) z2 += s_s[j] * uW2[j * HID + c];
    h_l[(size_t)l * HID + c] = hl_s[c] + z2;
}

// ---------------------------------------------------------------------------
// K5: mean-pool + v/w heads + rotations. One block, 128 threads.
// out[0:3] = R @ w_raw, out[3:6] = R @ v_raw.
// ---------------------------------------------------------------------------
__global__ void __launch_bounds__(128)
head_kernel(const float* __restrict__ h_l,
            const float* __restrict__ R,
            const float* __restrict__ vW1, const float* __restrict__ vb1,
            const float* __restrict__ vW2, const float* __restrict__ vb2,
            const float* __restrict__ wW1, const float* __restrict__ wb1,
            const float* __restrict__ wW2, const float* __restrict__ wb2,
            float* __restrict__ out) {
    __shared__ float pool[HID], vh[HID], wh[HID], vr[3], wr[3];
    const int j = threadIdx.x;
    float s = 0.0f;
    for (int l = 0; l < N_L; ++l) s += h_l[(size_t)l * HID + j];
    pool[j] = s * (1.0f / N_L);
    __syncthreads();
    float av = vb1[j], aw = wb1[j];
    for (int c = 0; c < HID; ++c) {
        av += pool[c] * vW1[c * HID + j];
        aw += pool[c] * wW1[c * HID + j];
    }
    vh[j] = silu_f(av);
    wh[j] = silu_f(aw);
    __syncthreads();
    if (j < 3) {
        float v = vb2[j], w = wb2[j];
        for (int k = 0; k < HID; ++k) {
            v += vh[k] * vW2[k * 3 + j];
            w += wh[k] * wW2[k * 3 + j];
        }
        vr[j] = v; wr[j] = w;
    }
    __syncthreads();
    if (j < 3) {
        float o1 = 0.0f, o2 = 0.0f;
#pragma unroll
        for (int k = 0; k < 3; ++k) {
            o1 += R[j * 3 + k] * wr[k];
            o2 += R[j * 3 + k] * vr[k];
        }
        out[j]     = o1;   // R @ w_raw
        out[3 + j] = o2;   // R @ v_raw
    }
}

// ---------------------------------------------------------------------------
extern "C" void kernel_launch(void* const* d_in, const int* in_sizes, int n_in,
                              void* d_out, int out_size, void* d_ws, size_t ws_size,
                              hipStream_t stream) {
    (void)in_sizes; (void)n_in; (void)out_size; (void)ws_size;
    auto P = [&](int i) { return (const float*)d_in[i]; };

    const float* h_l_in = P(0);
    const float* R_t    = P(1);
    const float* p_t    = P(2);
    const float* h_r_in = P(3);
    const float* x_r    = P(4);
    const float* t_in   = P(5);
    const float* embW   = P(6);
    const float* embB   = P(7);
    const float* tW1    = P(8);
    const float* tb1    = P(9);
    const float* tW2    = P(10);
    const float* tb2    = P(11);
    // heads
    const float* vW1 = P(42); const float* vb1 = P(43);
    const float* vW2 = P(44); const float* vb2 = P(45);
    const float* wW1 = P(46); const float* wb1 = P(47);
    const float* wW2 = P(48); const float* wb2 = P(49);

    float* ws = (float*)d_ws;
    float* ws_hl   = ws;                       // 256*128
    float* ws_hr   = ws + 32768;               // 2048*128
    float* ws_temb = ws + 294912;              // 128
    float* ws_xl   = ws + 295040;              // 256*3
    float* ws_pn   = ws + 295936;              // NSPLIT*256*128
    float* ws_pd   = ws + 295936 + NSPLIT * N_L * HID;

    time_mlp_kernel<<<1, 128, 0, stream>>>(t_in, tW1, tb1, tW2, tb2, ws_temb);
    embed_kernel<<<(N_L * HID) / 256, 256, 0, stream>>>(h_l_in, embW, embB, ws_temb, ws_hl);
    embed_kernel<<<(N_R * HID) / 256, 256, 0, stream>>>(h_r_in, embW, embB, nullptr, ws_hr);
    xl_kernel<<<1, 256, 0, stream>>>(ws_hl, R_t, p_t, ws_xl);

    for (int layer = 0; layer < 3; ++layer) {
        int base = 12 + layer * 10;
        const float* eW1 = P(base + 0);
        const float* eb1 = P(base + 1);
        const float* eW2 = P(base + 2);
        const float* eb2 = P(base + 3);
        const float* uW1 = P(base + 4);
        const float* ub1 = P(base + 5);
        const float* lng = P(base + 6);
        const float* lnb = P(base + 7);
        const float* uW2 = P(base + 8);
        const float* ub2 = P(base + 9);

        edge_agg_kernel<<<dim3(32, NSPLIT), 256, 0, stream>>>(
            ws_hl, ws_xl, ws_hr, x_r, eW1, eb1, eW2, eb2, ws_pn, ws_pd);
        update_kernel<<<N_L, 128, 0, stream>>>(
            ws_hl, ws_pn, ws_pd, uW1, ub1, lng, lnb, uW2, ub2);
    }

    head_kernel<<<1, 128, 0, stream>>>(ws_hl, R_t, vW1, vb1, vW2, vb2,
                                       wW1, wb1, wW2, wb2, (float*)d_out);
}
